// EnhancedUFormerBlock_69818988363882
// MI455X (gfx1250) — compile-verified, matrix operand order checked by reading, not run on hardware
//
#include <hip/hip_runtime.h>
#include <math.h>

typedef __attribute__((ext_vector_type(16))) _Float16 v16h;
typedef __attribute__((ext_vector_type(8)))  float    v8f;

#define DIM 128
#define HALO 8
#define HEADS 8
#define HIDDEN 512
#define NPIX 16384   // 128*128

// ---------------- fragment load helpers ----------------
union FragU { v16h v; uint4 u[2]; };

__device__ inline v16h frag_full(const _Float16* p0, const _Float16* p1) {
    FragU t;
    t.u[0] = *(const uint4*)p0;
    t.u[1] = *(const uint4*)p1;
    return t.v;
}
__device__ inline v16h frag_low(const _Float16* p0) {
    FragU t;
    t.u[0] = *(const uint4*)p0;
    t.u[1] = make_uint4(0u, 0u, 0u, 0u);
    return t.v;
}
__device__ inline v8f wmma_f16(v16h a, v16h b, v8f c) {
    return __builtin_amdgcn_wmma_f32_16x16x32_f16(false, a, false, b, (short)0, c, false, false);
}

// ---------------- utility kernels ----------------
__global__ void k_chw2hwc(const float* __restrict__ x, float* __restrict__ xh) {
    int idx = blockIdx.x * 256 + threadIdx.x;          // 2,097,152 elems
    int c = idx >> 14;
    int p = idx & 16383;
    xh[(size_t)p * DIM + c] = x[idx];
}

__global__ void k_hwc2chw(const float* __restrict__ xh, float* __restrict__ out) {
    int idx = blockIdx.x * 256 + threadIdx.x;
    int c = idx >> 14;
    int p = idx & 16383;
    out[idx] = xh[(size_t)p * DIM + c];
}

__global__ void k_cvt_f16(const float* __restrict__ src, _Float16* __restrict__ dst, int n) {
    int i = blockIdx.x * 256 + threadIdx.x;
    if (i < n) dst[i] = (_Float16)src[i];
}

// LayerNorm over C=128; one wave per pixel, 8 pixels per block.
__global__ void k_layernorm(const float* __restrict__ in, const float* __restrict__ w,
                            const float* __restrict__ b, _Float16* __restrict__ out) {
    int wave = threadIdx.x >> 5;
    int lane = threadIdx.x & 31;
    int pixel = blockIdx.x * 8 + wave;
    const float* p = in + (size_t)pixel * DIM;
    float4 v = *(const float4*)(p + lane * 4);
    float s  = v.x + v.y + v.z + v.w;
    float ss = v.x * v.x + v.y * v.y + v.z * v.z + v.w * v.w;
    #pragma unroll
    for (int m = 1; m < 32; m <<= 1) {
        s  += __shfl_xor(s,  m, 32);
        ss += __shfl_xor(ss, m, 32);
    }
    float mu  = s * (1.0f / 128.0f);
    float var = ss * (1.0f / 128.0f) - mu * mu;
    float inv = rsqrtf(var + 1e-5f);
    int c = lane * 4;
    _Float16* o = out + (size_t)pixel * DIM + c;
    o[0] = (_Float16)((v.x - mu) * inv * w[c + 0] + b[c + 0]);
    o[1] = (_Float16)((v.y - mu) * inv * w[c + 1] + b[c + 1]);
    o[2] = (_Float16)((v.z - mu) * inv * w[c + 2] + b[c + 2]);
    o[3] = (_Float16)((v.w - mu) * inv * w[c + 3] + b[c + 3]);
}

// ---------------- WMMA GEMM: Y = X[M,K] * W[N,K]^T + bias ----------------
// Each wave computes a 16x64 output strip (4 accumulators, A-frag reused 4x).
// mode 0: -> f16 out.  mode 1: out_f32 = resid + Y.  mode 2: gelu(Y) -> f16 out.
__global__ void k_gemm(const _Float16* __restrict__ X, const _Float16* __restrict__ W,
                       const float* __restrict__ bias, const float* __restrict__ resid,
                       float* __restrict__ outf, _Float16* __restrict__ outh,
                       int K, int N, int mode) {
    int wave = threadIdx.x >> 5;
    int lane = threadIdx.x & 31;
    int g = lane >> 4;          // half-wave group
    int cw = lane & 15;
    int row0 = blockIdx.x * 128 + wave * 16;
    int col0 = blockIdx.y * 64;

    v8f acc[4];
    #pragma unroll
    for (int j = 0; j < 4; ++j) acc[j] = (v8f){0.f,0.f,0.f,0.f,0.f,0.f,0.f,0.f};

    const _Float16* xrow = X + (size_t)(row0 + cw) * K + g * 8;
    const _Float16* wrow = W + (size_t)(col0 + cw) * K + g * 8;
    for (int k0 = 0; k0 < K; k0 += 32) {
        v16h a = frag_full(xrow + k0, xrow + k0 + 16);
        #pragma unroll
        for (int j = 0; j < 4; ++j) {
            const _Float16* wp = wrow + (size_t)j * 16 * K + k0;
            v16h b = frag_full(wp, wp + 16);
            acc[j] = wmma_f16(a, b, acc[j]);
        }
    }
    #pragma unroll
    for (int j = 0; j < 4; ++j) {
        #pragma unroll
        for (int r = 0; r < 8; ++r) {
            int row = row0 + r + 8 * g;
            int col = col0 + j * 16 + cw;
            float v = acc[j][r] + bias[col];
            size_t o = (size_t)row * N + col;
            if (mode == 1) {
                outf[o] = resid[o] + v;
            } else {
                if (mode == 2) v = 0.5f * v * (1.0f + erff(v * 0.70710678118f));
                outh[o] = (_Float16)v;
            }
        }
    }
}

// ---------------- halo attention (flash-style online softmax) ----------------
// grid: (64 windows, 8 heads), block 256 threads (8 waves, 2 query tiles each)
__global__ void k_halo_attn(const _Float16* __restrict__ qkv, _Float16* __restrict__ attn) {
    __shared__ __align__(16) _Float16 qs[256 * 16];        // scaled q  [qi][d]
    __shared__ __align__(16) _Float16 ks[64 * 16];         // k chunk   [kl][d]
    __shared__ __align__(16) _Float16 vts[16 * 64];        // v chunk T [d][kl]
    __shared__ __align__(16) _Float16 ps[8 * 2 * 16 * 64]; // P per wave/tile
    __shared__ float m_s[256];
    __shared__ float l_s[256];

    int tid  = threadIdx.x;
    int wave = tid >> 5;
    int lane = tid & 31;
    int g    = lane >> 4;
    int cw   = lane & 15;
    int wy = blockIdx.x >> 3, wx = blockIdx.x & 7;
    int head = blockIdx.y;

    // stage q (scaled by HD^-0.5 = 0.25), init stats
    {
        int qi = tid;
        int py = qi >> 4, px = qi & 15;
        size_t gp = (size_t)((wy * 16 + py) * 128 + (wx * 16 + px));
        const _Float16* src = qkv + gp * 384 + head * 16;
        #pragma unroll
        for (int d = 0; d < 16; ++d)
            qs[qi * 16 + d] = (_Float16)((float)src[d] * 0.25f);
        m_s[qi] = -1e30f;
        l_s[qi] = 0.0f;
    }

    v8f acc[2];
    acc[0] = (v8f){0.f,0.f,0.f,0.f,0.f,0.f,0.f,0.f};
    acc[1] = (v8f){0.f,0.f,0.f,0.f,0.f,0.f,0.f,0.f};

    for (int ch = 0; ch < 16; ++ch) {      // 16 chunks of 64 keys = 1024
        __syncthreads();
        // ---- stage k chunk: CDNA5 async global->LDS copies (ASYNCcnt) ----
        if (tid < 128) {
            int key  = tid >> 1;            // 0..63
            int half = tid & 1;             // 16B half of the 32B row
            int kk = ch * 64 + key;
            int ky = kk >> 5, kx = kk & 31;
            int gy = wy * 16 - HALO + ky;
            int gx = wx * 16 - HALO + kx;
            bool ok = (gy >= 0) & (gy < 128) & (gx >= 0) & (gx < 128);
            _Float16* ldst = ks + key * 16 + half * 8;
            if (ok) {
                const _Float16* src =
                    qkv + (size_t)(gy * 128 + gx) * 384 + 128 + head * 16 + half * 8;
                unsigned lo = (unsigned)(unsigned long long)(void*)ldst;
                unsigned long long ga = (unsigned long long)(const void*)src;
                asm volatile("global_load_async_to_lds_b128 %0, %1, off"
                             :: "v"(lo), "v"(ga) : "memory");
            } else {
                #pragma unroll
                for (int j = 0; j < 8; ++j) ldst[j] = (_Float16)0.0f;
            }
        }
        // ---- stage v chunk transposed (normal path; async can't transpose) ----
        if (tid < 64) {
            int kk = ch * 64 + tid;
            int ky = kk >> 5, kx = kk & 31;
            int gy = wy * 16 - HALO + ky;
            int gx = wx * 16 - HALO + kx;
            bool ok = (gy >= 0) & (gy < 128) & (gx >= 0) & (gx < 128);
            size_t gp = ok ? (size_t)(gy * 128 + gx) : 0;
            const _Float16* vp = qkv + gp * 384 + 256 + head * 16;
            #pragma unroll
            for (int d = 0; d < 16; ++d)
                vts[d * 64 + tid] = ok ? vp[d] : (_Float16)0.0f;
        }
        asm volatile("s_wait_asynccnt 0x0" ::: "memory");
        __syncthreads();

        #pragma unroll
        for (int t2 = 0; t2 < 2; ++t2) {
            int tile = wave * 2 + t2;
            // A fragment from q (upper 16 K zero-padded)
            v16h a = frag_low(qs + (tile * 16 + cw) * 16 + g * 8);
            v8f s[4];
            #pragma unroll
            for (int kt = 0; kt < 4; ++kt) {
                v16h b = frag_low(ks + (kt * 16 + cw) * 16 + g * 8);
                v8f z = {0.f,0.f,0.f,0.f,0.f,0.f,0.f,0.f};
                s[kt] = wmma_f16(a, b, z);
            }
            _Float16* pbase = ps + (wave * 2 + t2) * (16 * 64);
            #pragma unroll
            for (int r = 0; r < 8; ++r) {
                float cm = fmaxf(fmaxf(s[0][r], s[1][r]), fmaxf(s[2][r], s[3][r]));
                #pragma unroll
                for (int m = 1; m < 16; m <<= 1) cm = fmaxf(cm, __shfl_xor(cm, m, 32));
                int row = tile * 16 + r + 8 * g;
                float mo = m_s[row];
                float mn = fmaxf(mo, cm);
                float scl = __expf(mo - mn);
                float rs = 0.0f;
                #pragma unroll
                for (int kt = 0; kt < 4; ++kt) {
                    float pv = __expf(s[kt][r] - mn);
                    s[kt][r] = pv;
                    rs += pv;
                }
                #pragma unroll
                for (int m = 1; m < 16; m <<= 1) rs += __shfl_xor(rs, m, 32);
                if (cw == 0) { m_s[row] = mn; l_s[row] = l_s[row] * scl + rs; }
                acc[t2][r] *= scl;
                int prow = r + 8 * g;
                #pragma unroll
                for (int kt = 0; kt < 4; ++kt)
                    pbase[prow * 64 + kt * 16 + cw] = (_Float16)s[kt][r];
            }
            // P @ V : K = 64 keys -> two 16x16x32 wmma
            #pragma unroll
            for (int kk2 = 0; kk2 < 2; ++kk2) {
                const _Float16* pa = pbase + cw * 64 + kk2 * 32 + g * 8;
                v16h af = frag_full(pa, pa + 16);
                const _Float16* pb = vts + cw * 64 + kk2 * 32 + g * 8;
                v16h bf = frag_full(pb, pb + 16);
                acc[t2] = wmma_f16(af, bf, acc[t2]);
            }
        }
    }
    __syncthreads();

    #pragma unroll
    for (int t2 = 0; t2 < 2; ++t2) {
        int tile = wave * 2 + t2;
        #pragma unroll
        for (int r = 0; r < 8; ++r) {
            int row = tile * 16 + r + 8 * g;
            float val = acc[t2][r] / l_s[row];
            int py = row >> 4, px = row & 15;
            size_t gp = (size_t)((wy * 16 + py) * 128 + (wx * 16 + px));
            attn[gp * 128 + head * 16 + cw] = (_Float16)val;
        }
    }
}

// ---------------- launcher ----------------
extern "C" void kernel_launch(void* const* d_in, const int* in_sizes, int n_in,
                              void* d_out, int out_size, void* d_ws, size_t ws_size,
                              hipStream_t stream) {
    const float* x      = (const float*)d_in[0];
    const float* ln1_w  = (const float*)d_in[1];
    const float* ln1_b  = (const float*)d_in[2];
    const float* qkv_w  = (const float*)d_in[3];
    const float* qkv_b  = (const float*)d_in[4];
    const float* proj_w = (const float*)d_in[5];
    const float* proj_b = (const float*)d_in[6];
    const float* ln2_w  = (const float*)d_in[7];
    const float* ln2_b  = (const float*)d_in[8];
    const float* fc1_w  = (const float*)d_in[9];
    const float* fc1_b  = (const float*)d_in[10];
    const float* fc2_w  = (const float*)d_in[11];
    const float* fc2_b  = (const float*)d_in[12];
    float* out = (float*)d_out;

    char* ws = (char*)d_ws;
    size_t off = 0;
    auto alloc = [&](size_t bytes) { char* p = ws + off; off += (bytes + 255) & ~(size_t)255; return p; };
    float*     xh   = (float*)alloc((size_t)NPIX * DIM * 4);
    float*     xh2  = (float*)alloc((size_t)NPIX * DIM * 4);
    float*     xh3  = (float*)alloc((size_t)NPIX * DIM * 4);
    _Float16*  hln1 = (_Float16*)alloc((size_t)NPIX * DIM * 2);
    _Float16*  hln2 = (_Float16*)alloc((size_t)NPIX * DIM * 2);
    _Float16*  qkvb = (_Float16*)alloc((size_t)NPIX * 384 * 2);
    _Float16*  attn = (_Float16*)alloc((size_t)NPIX * DIM * 2);
    _Float16*  mlp1 = (_Float16*)alloc((size_t)NPIX * HIDDEN * 2);
    _Float16*  wq   = (_Float16*)alloc((size_t)384 * DIM * 2);
    _Float16*  wp   = (_Float16*)alloc((size_t)DIM * DIM * 2);
    _Float16*  w1   = (_Float16*)alloc((size_t)HIDDEN * DIM * 2);
    _Float16*  w2   = (_Float16*)alloc((size_t)DIM * HIDDEN * 2);

    const int ELN = NPIX * DIM;   // 2,097,152

    k_chw2hwc<<<ELN / 256, 256, 0, stream>>>(x, xh);
    k_cvt_f16<<<(384 * DIM + 255) / 256, 256, 0, stream>>>(qkv_w, wq, 384 * DIM);
    k_cvt_f16<<<(DIM * DIM + 255) / 256, 256, 0, stream>>>(proj_w, wp, DIM * DIM);
    k_cvt_f16<<<(HIDDEN * DIM + 255) / 256, 256, 0, stream>>>(fc1_w, w1, HIDDEN * DIM);
    k_cvt_f16<<<(DIM * HIDDEN + 255) / 256, 256, 0, stream>>>(fc2_w, w2, DIM * HIDDEN);

    k_layernorm<<<NPIX / 8, 256, 0, stream>>>(xh, ln1_w, ln1_b, hln1);

    // QKV: [16384,128] x [384,128]^T -> f16
    k_gemm<<<dim3(NPIX / 128, 384 / 64), 256, 0, stream>>>(hln1, wq, qkv_b, nullptr,
                                                           nullptr, qkvb, DIM, 384, 0);
    // halo attention
    k_halo_attn<<<dim3(64, HEADS), 256, 0, stream>>>(qkvb, attn);

    // proj + residual -> xh2 (f32)
    k_gemm<<<dim3(NPIX / 128, DIM / 64), 256, 0, stream>>>(attn, wp, proj_b, xh,
                                                           xh2, nullptr, DIM, DIM, 1);
    k_layernorm<<<NPIX / 8, 256, 0, stream>>>(xh2, ln2_w, ln2_b, hln2);

    // fc1 + gelu -> f16
    k_gemm<<<dim3(NPIX / 128, HIDDEN / 64), 256, 0, stream>>>(hln2, w1, fc1_b, nullptr,
                                                              nullptr, mlp1, DIM, HIDDEN, 2);
    // fc2 + residual -> xh3 (f32)
    k_gemm<<<dim3(NPIX / 128, DIM / 64), 256, 0, stream>>>(mlp1, w2, fc2_b, xh2,
                                                           xh3, nullptr, HIDDEN, DIM, 1);

    k_hwc2chw<<<ELN / 256, 256, 0, stream>>>(xh3, out);
}
